// HiPPOLegSBlock_57870389346870
// MI455X (gfx1250) — compile-verified, matrix operand order checked by reading, not run on hardware
//
#include <hip/hip_runtime.h>
#include <stdint.h>

typedef __attribute__((ext_vector_type(16))) _Float16 v16h;
typedef __attribute__((ext_vector_type(8)))  float    v8f;
typedef int v2i __attribute__((vector_size(8)));     // b64 payload for async-LDS

#define B_   8
#define T_   32768
#define FIN  64
#define HID  32
#define NST  64
#define M_   (B_*T_)       // 262144 rows of the fused (B*T) x FIN matrix
#define LCH  128           // scan chunk length
#define NCH  (T_/LCH)      // 256 chunks

// ---- workspace layout (float offsets) ------------------------------------
#define WS_U      0                         // u[b,t]           M_ floats
#define WS_DT     (WS_U   + M_)             // dterm[b,t]       M_ floats
#define WS_E      (WS_DT  + M_)             // chunk endpoints  B_*NCH*NST
#define WS_CIN    (WS_E   + B_*NCH*NST)     // chunk carry-ins  B_*NCH*NST
#define WS_A      (WS_CIN + B_*NCH*NST)     // A[n]   (64)
#define WS_CP     (WS_A   + 64)             // C[n]*P[n]
#define WS_AL     (WS_CP  + 64)             // A[n]^LCH
#define WS_V1     (WS_AL  + 64)             // W2 @ Wu (32)
#define WS_V2     (WS_V1  + 32)             // W2 @ D  (32)
#define WS_C12    (WS_V2  + 32)             // c1=b2.Wu, c2=b2.D (+pad)
#define WS_WPACK  (WS_C12 + 32)             // W1 packed f16 pairs, 1024 u32
// total ~3.2 MB

// ---- gfx1250 async load-to-LDS (ASYNCcnt path), guarded ------------------
#if __has_builtin(__builtin_amdgcn_global_load_async_to_lds_b64) && \
    __has_builtin(__builtin_amdgcn_s_wait_asynccnt)
#define HAVE_ASYNC_LDS 1
#else
#define HAVE_ASYNC_LDS 0
#endif

// Stage one LCH-float chunk of u into LDS with 64 threads.
// Caller must __syncthreads() afterwards.
__device__ __forceinline__ void stage_u_chunk(const float* __restrict__ u,
                                              float* su, int n) {
#if HAVE_ASYNC_LDS
  // each thread: one async B64 (2 floats) straight into LDS, no VGPR round-trip
  __builtin_amdgcn_global_load_async_to_lds_b64(
      (__attribute__((address_space(1))) v2i*)(u + 2 * n),
      (__attribute__((address_space(3))) v2i*)(su + 2 * n),
      0, 0);
  __builtin_amdgcn_s_wait_asynccnt(0);
#else
  su[n] = u[n];
  su[n + 64] = u[n + 64];
#endif
}

// ---- prep: SSM constants, folded vectors, W1 -> WMMA B-operand layout ----
__global__ void prep_kernel(const float* __restrict__ W1, const float* __restrict__ W2,
                            const float* __restrict__ b2, const float* __restrict__ Lraw,
                            const float* __restrict__ P,  const float* __restrict__ Wu,
                            const float* __restrict__ C,  const float* __restrict__ D,
                            float* __restrict__ ws) {
  int tid = threadIdx.x;                       // 256 threads, 1 block
  if (tid < NST) {
    float xr = Lraw[tid];
    float sp = (xr > 20.f) ? xr : log1pf(expf(xr));   // softplus, stable
    ws[WS_A  + tid] = expf(-sp);                      // A = exp(-softplus)
    ws[WS_AL + tid] = expf(-(float)LCH * sp);         // A^LCH
    ws[WS_CP + tid] = C[tid] * P[tid];
  }
  if (tid < HID) {
    float s1 = 0.f, s2 = 0.f;
    for (int f = 0; f < FIN; ++f) {
      float w = W2[tid * FIN + f];
      s1 = fmaf(w, Wu[f], s1);
      s2 = fmaf(w, D[f],  s2);
    }
    ws[WS_V1 + tid] = s1;
    ws[WS_V2 + tid] = s2;
  }
  if (tid == 0) {
    float c1 = 0.f, c2 = 0.f;
    for (int f = 0; f < FIN; ++f) { c1 = fmaf(b2[f], Wu[f], c1); c2 = fmaf(b2[f], D[f], c2); }
    ws[WS_C12]     = c1;
    ws[WS_C12 + 1] = c2;
  }
  // Pack W1 (FIN x HID, row-major, K x N) into the 16-bit B-operand lane layout
  // (CDNA5 ISA 7.12.2): lane<16 owns col n=lane with K pairs
  // {0,1..6,7}U{16..23}; lane>=16 owns same col set shifted by K+8.
  uint32_t* wp = (uint32_t*)(ws + WS_WPACK);
  for (int idx = tid; idx < 1024; idx += 256) {
    int s    = idx & 7;
    int lane = (idx >> 3) & 31;
    int nt   = (idx >> 8) & 1;
    int kt   = idx >> 9;
    int k0 = kt * 32 + ((lane >= 16) ? 8 : 0) + ((s < 4) ? 2 * s : 16 + 2 * (s - 4));
    int n  = nt * 16 + (lane & 15);
    union { _Float16 h; uint16_t u; } lo, hi;
    lo.h = (_Float16)W1[(size_t)k0 * HID + n];
    hi.h = (_Float16)W1[(size_t)(k0 + 1) * HID + n];
    wp[idx] = (uint32_t)lo.u | ((uint32_t)hi.u << 16);
  }
}

// ---- featurize: WMMA GEMM (M x 64)@(64 x 32), relu, fold to u & dterm ----
__global__ void __launch_bounds__(256) feat_kernel(const float* __restrict__ x,
                                                   const float* __restrict__ b1,
                                                   float* __restrict__ ws) {
  __shared__ float Hs[8][16][33];              // per-wave 16x32 relu'd H tile (+pad)
  const int lane = threadIdx.x & 31;
  const int w    = threadIdx.x >> 5;           // 8 waves / block, one 16-row tile each
  const int tile = blockIdx.x * 8 + w;
  const int m0   = tile * 16;
  const int row  = m0 + (lane & 15);

  // B operand: pre-packed W1, 2 uint4 per lane per (kt,nt)
  const uint4* wp = (const uint4*)(ws + WS_WPACK);
  v16h bmat[2][2];
#pragma unroll
  for (int kt = 0; kt < 2; ++kt)
#pragma unroll
    for (int nt = 0; nt < 2; ++nt) {
      union { uint4 q[2]; v16h v; } tmp;
      int base = ((kt * 2 + nt) * 32 + lane) * 2;
      tmp.q[0] = wp[base];
      tmp.q[1] = wp[base + 1];
      bmat[kt][nt] = tmp.v;
    }

  const float* xr = x + (size_t)row * FIN;
  __builtin_prefetch(xr + 128 * FIN, 0, 1);    // global_prefetch_b8 on streaming x

  v8f acc0 = {}; v8f acc1 = {};
  const int kb = (lane >= 16) ? 8 : 0;         // A-operand K base per lane half
#pragma unroll
  for (int kt = 0; kt < 2; ++kt) {
    const float4* q = (const float4*)(xr + kt * 32 + kb);
    float4 q0 = q[0], q1 = q[1], q2 = q[4], q3 = q[5];  // K: +0..7 and +16..23
    v16h a;
    a[0]  = (_Float16)q0.x; a[1]  = (_Float16)q0.y; a[2]  = (_Float16)q0.z; a[3]  = (_Float16)q0.w;
    a[4]  = (_Float16)q1.x; a[5]  = (_Float16)q1.y; a[6]  = (_Float16)q1.z; a[7]  = (_Float16)q1.w;
    a[8]  = (_Float16)q2.x; a[9]  = (_Float16)q2.y; a[10] = (_Float16)q2.z; a[11] = (_Float16)q2.w;
    a[12] = (_Float16)q3.x; a[13] = (_Float16)q3.y; a[14] = (_Float16)q3.z; a[15] = (_Float16)q3.w;
    acc0 = __builtin_amdgcn_wmma_f32_16x16x32_f16(false, a, false, bmat[kt][0],
                                                  (short)0, acc0, false, false);
    acc1 = __builtin_amdgcn_wmma_f32_16x16x32_f16(false, a, false, bmat[kt][1],
                                                  (short)0, acc1, false, false);
  }

  // bias + relu, scatter C-layout (col=lane%16, rows in VGPRs) into LDS
  const int cb = lane & 15;
  const int rb = (lane < 16) ? 0 : 8;
  const float bn0 = b1[cb];
  const float bn1 = b1[16 + cb];
#pragma unroll
  for (int v = 0; v < 8; ++v) {
    Hs[w][rb + v][cb]      = fmaxf(acc0[v] + bn0, 0.f);
    Hs[w][rb + v][16 + cb] = fmaxf(acc1[v] + bn1, 0.f);
  }
  __syncthreads();

  // fold: lanes 0..15 -> u = H.v1 + c1 ; lanes 16..31 -> dterm = H.v2 + c2
  const int half = lane >> 4;
  const int r    = lane & 15;
  const float* vv = ws + (half ? WS_V2 : WS_V1);
  float acc = ws[WS_C12 + half];
#pragma unroll
  for (int h = 0; h < HID; ++h) acc = fmaf(Hs[w][r][h], vv[h], acc);
  float* dst = ws + (half ? WS_DT : WS_U);
  dst[m0 + r] = acc;
}

// ---- phase A: zero-init endpoint of each chunk's scalar scan -------------
__global__ void __launch_bounds__(64) scanA_kernel(float* __restrict__ ws) {
  __shared__ float su[LCH];
  const int b = blockIdx.x / NCH, c = blockIdx.x % NCH;
  const int n = threadIdx.x;
  const float* u = ws + WS_U + (size_t)b * T_ + (size_t)c * LCH;
  stage_u_chunk(u, su, n);
  __syncthreads();
  const float An = ws[WS_A + n];
  float g = 0.f;
#pragma unroll 8
  for (int j = 0; j < LCH; ++j) g = fmaf(An, g, su[j]);
  ws[WS_E + ((size_t)b * NCH + c) * NST + n] = g;
}

// ---- phase B: compose carries across chunks ------------------------------
__global__ void __launch_bounds__(512) scanB_kernel(float* __restrict__ ws) {
  const int tid = threadIdx.x;                 // 512 = B_*NST
  const int b = tid >> 6, n = tid & 63;
  const float ALn = ws[WS_AL + n];
  float carry = 0.f;
  for (int c = 0; c < NCH; ++c) {
    size_t i = ((size_t)b * NCH + c) * NST + n;
    ws[WS_CIN + i] = carry;
    carry = fmaf(ALn, carry, ws[WS_E + i]);
  }
}

// ---- phase C: rescan with carry, reduce over N via LDS tile, emit y ------
__global__ void __launch_bounds__(64) scanC_kernel(const float* __restrict__ by,
                                                   float* __restrict__ ws,
                                                   float* __restrict__ out) {
  __shared__ float su[LCH];
  __shared__ float Gt[NST][LCH + 1];           // +1 pad vs bank conflicts
  const int b = blockIdx.x / NCH, c = blockIdx.x % NCH;
  const int n = threadIdx.x;
  const float* u = ws + WS_U + (size_t)b * T_ + (size_t)c * LCH;
  stage_u_chunk(u, su, n);
  __syncthreads();
  const float An  = ws[WS_A + n];
  const float CPn = ws[WS_CP + n];
  float g = ws[WS_CIN + ((size_t)b * NCH + c) * NST + n];
  for (int j = 0; j < LCH; ++j) {
    g = fmaf(An, g, su[j]);
    Gt[n][j] = CPn * g;
  }
  __syncthreads();
  const float bias = by[0];
  const float* dt = ws + WS_DT + (size_t)b * T_ + (size_t)c * LCH;
  for (int j = n; j < LCH; j += 64) {
    float acc = 0.f;
#pragma unroll 8
    for (int k = 0; k < NST; ++k) acc += Gt[k][j];
    out[(size_t)b * T_ + (size_t)c * LCH + j] = acc + dt[j] + bias;
  }
}

extern "C" void kernel_launch(void* const* d_in, const int* in_sizes, int n_in,
                              void* d_out, int out_size, void* d_ws, size_t ws_size,
                              hipStream_t stream) {
  (void)in_sizes; (void)n_in; (void)out_size; (void)ws_size;
  const float* x  = (const float*)d_in[0];
  const float* W1 = (const float*)d_in[1];
  const float* b1 = (const float*)d_in[2];
  const float* W2 = (const float*)d_in[3];
  const float* b2 = (const float*)d_in[4];
  const float* Lr = (const float*)d_in[5];
  const float* P  = (const float*)d_in[6];
  const float* Wu = (const float*)d_in[7];
  const float* C  = (const float*)d_in[8];
  const float* D  = (const float*)d_in[9];
  const float* by = (const float*)d_in[10];
  float* out = (float*)d_out;
  float* ws  = (float*)d_ws;

  prep_kernel <<<1,           256, 0, stream>>>(W1, W2, b2, Lr, P, Wu, C, D, ws);
  feat_kernel <<<M_ / 128,    256, 0, stream>>>(x, b1, ws);
  scanA_kernel<<<B_ * NCH,     64, 0, stream>>>(ws);
  scanB_kernel<<<1,           512, 0, stream>>>(ws);
  scanC_kernel<<<B_ * NCH,     64, 0, stream>>>(by, ws, out);
}
